// BranchingLayer_47253230191193
// MI455X (gfx1250) — matrix operand order repaired and checked by reading, not compiled, not run on hardware
//
#include <hip/hip_runtime.h>
#include <hip/hip_bf16.h>

// Problem constants
#define NE 131072
#define NF 128
#define NG 64
#define NH 512

typedef __attribute__((ext_vector_type(16))) __bf16 v16bf;
typedef __attribute__((ext_vector_type(8)))  __bf16 v8bf;
typedef __attribute__((ext_vector_type(2)))  __bf16 v2bf;
typedef __attribute__((ext_vector_type(8)))  float  v8f;
typedef __attribute__((ext_vector_type(4)))  unsigned int u32x4;
typedef __attribute__((ext_vector_type(8)))  int i32x8;
typedef __attribute__((ext_vector_type(4)))  int i32x4;

// Workspace layout: swizzled bf16 weights
#define W1S_BYTES (192 * 512 * 2)   // 196608: tiles (nt-major): t = nt*6 + kt, 1KB each
#define W2S_BYTES (512 * 256 * 2)   // 262144: tiles (kt-major): t = kt*16 + nt, 1KB each
#define W1_DW (W1S_BYTES / 4)       // 49152
#define W2_DW (W2S_BYTES / 4)       // 65536

// LDS layout
#define HS_STRIDE 520                        // bf16 elems per row (512 + 8 pad), 1040 B
#define HS_BYTES (128 * HS_STRIDE * 2)       // 133120
#define WSTAGE_BYTES 49152
#define SMEM_BYTES (HS_BYTES + WSTAGE_BYTES) // 182272

#if defined(__HIP_DEVICE_COMPILE__) && __has_builtin(__builtin_amdgcn_tensor_load_to_lds)
#define HAS_TDM 1
#else
#define HAS_TDM 0
#endif

static __device__ __forceinline__ v16bf concat8(v8bf a, v8bf b) {
  return __builtin_shufflevector(a, b, 0,1,2,3,4,5,6,7,8,9,10,11,12,13,14,15);
}

#if HAS_TDM
// Stage `nbytes` (multiple of 8, contiguous) from global -> LDS via the Tensor
// Data Mover. D# per CDNA5 ISA §8: group0 = {ctrl, lds_addr, gaddr_lo, gaddr_hi|type},
// group1 = dims/strides (data_size=8B, one row of nbytes/8 elements).
static __device__ __forceinline__ void tdm_stage(unsigned lds_off, const void* gsrc,
                                                 unsigned nbytes) {
  unsigned long long ga = (unsigned long long)(size_t)gsrc;
  unsigned n8 = nbytes >> 3;  // elements of 8 bytes (fits 16-bit tile_dim0)
  u32x4 g0;
  g0[0] = 1u;                                                  // count=1, user D#
  g0[1] = lds_off;                                             // LDS byte address
  g0[2] = (unsigned)ga;                                        // global addr [31:0]
  g0[3] = (unsigned)((ga >> 32) & 0x1FFFFFFu) | (2u << 30);    // addr[56:32] | type=2
  i32x8 g1;
  g1[0] = (int)(3u << 16);                 // data_size = 8 bytes
  g1[1] = (int)((n8 & 0xFFFFu) << 16);     // tensor_dim0[15:0]
  g1[2] = (int)((n8 >> 16) | (1u << 16));  // tensor_dim0[31:16], tensor_dim1 = 1
  g1[3] = (int)(((unsigned)n8 & 0xFFFFu) << 16);  // tile_dim0 = n8
  g1[4] = 1;                               // tile_dim1 = 1, tile_dim2 = 0
  g1[5] = (int)n8;                         // tensor_dim0_stride[31:0]
  g1[6] = 0;
  g1[7] = 0;
  i32x4 z4 = {0, 0, 0, 0};
#if __clang_major__ >= 23
  i32x8 z8 = {0, 0, 0, 0, 0, 0, 0, 0};
  __builtin_amdgcn_tensor_load_to_lds(g0, g1, z4, z4, z8, 0);
#else
  __builtin_amdgcn_tensor_load_to_lds(g0, g1, z4, z4, 0);
#endif
}
#endif

// ---------------------------------------------------------------------------
// Prep: convert W1/W2 fp32 -> bf16 in B-operand swizzled tile layout.
// Tile (32K x 16N) = 1024 B: lane L (0..31) holds 32 bytes = 16 consecutive K
// for column n = 16*nt + (L&15), starting at kbase = 32*kt + (L<16 ? 0 : 16).
// ---------------------------------------------------------------------------
__global__ __launch_bounds__(256) void prep_weights(
    const float* __restrict__ W1, const float* __restrict__ W2,
    v2bf* __restrict__ w1s, v2bf* __restrict__ w2s) {
  int d = blockIdx.x * 256 + threadIdx.x;
  if (d < W1_DW) {
    int t = d >> 8, r = d & 255, lane = r >> 3, q = r & 7;
    int nt = t / 6, kt = t % 6;
    int n = nt * 16 + (lane & 15);
    int k = kt * 32 + ((lane < 16) ? 0 : 16) + q * 2;
    v2bf v;
    v[0] = (__bf16)W1[k * NH + n];
    v[1] = (__bf16)W1[(k + 1) * NH + n];
    w1s[d] = v;
  } else if (d < W1_DW + W2_DW) {
    int d2 = d - W1_DW;
    int t = d2 >> 8, r = d2 & 255, lane = r >> 3, q = r & 7;
    int kt = t >> 4, nt = t & 15;
    int n = nt * 16 + (lane & 15);
    int k = kt * 32 + ((lane < 16) ? 0 : 16) + q * 2;
    v2bf v;
    v[0] = (__bf16)W2[k * 256 + n];
    v[1] = (__bf16)W2[(k + 1) * 256 + n];
    w2s[d2] = v;
  }
}

// ---------------------------------------------------------------------------
// Copy x -> out[0 : NE*NF] (coalesced float4)
// ---------------------------------------------------------------------------
__global__ __launch_bounds__(256) void copy_x(const float4* __restrict__ src,
                                              float4* __restrict__ dst) {
  unsigned i = blockIdx.x * 256u + threadIdx.x;  // 16384 blocks * 256 = NE*NF/4
  dst[i] = src[i];
}

// ---------------------------------------------------------------------------
// Fused: h = leaky_relu([x|g] @ W1 + b1);  proj = h @ W2 + b2; branch-scatter.
// 1024 WGs x 256 threads (8 waves). Wave w owns local event rows 16w..16w+15.
// ---------------------------------------------------------------------------
__global__ __launch_bounds__(256) void fused_branching(
    const float* __restrict__ x, const float* __restrict__ g,
    const float* __restrict__ b1, const float* __restrict__ b2,
    const __bf16* __restrict__ w1s, const __bf16* __restrict__ w2s,
    const int* __restrict__ pidx, float* __restrict__ out) {
  extern __shared__ char smem[];
  __bf16* Hs = (__bf16*)smem;               // [128][HS_STRIDE] bf16
  char* Wstage = smem + HS_BYTES;           // 48 KB weight staging

  const int tid  = threadIdx.x;
  const int lane = tid & 31;
  const int w    = tid >> 5;        // wave 0..7
  const int half = lane >> 4;       // 0|1
  const int lm   = lane & 15;
  const int klo  = half * 16;
  const int m_base = w * 16;        // local row slab base

#if HAS_TDM
  const unsigned wstage_lds_off =
      (unsigned)__builtin_amdgcn_groupstaticsize() + (unsigned)HS_BYTES;
#endif

  // ---- Stage A: A fragments for this wave's 16 rows (K = 192 -> 6 frags) ----
  const int e_row = blockIdx.x * 128 + m_base + lm;      // destination row
  const unsigned esrc = (unsigned)pidx[e_row];           // parent gather
  const unsigned eg = esrc & (NE - 1);                   // % n_events

  v16bf Af[6];
#pragma unroll
  for (int kt = 0; kt < 6; ++kt) {
    const float* p = (kt < 4) ? (x + (size_t)esrc * NF + kt * 32 + klo)
                              : (g + (size_t)eg * NG + (kt - 4) * 32 + klo);
    float4 a0 = *(const float4*)(p + 0);
    float4 a1 = *(const float4*)(p + 4);
    float4 a2 = *(const float4*)(p + 8);
    float4 a3 = *(const float4*)(p + 12);
    v16bf f;
    f[0]  = (__bf16)a0.x; f[1]  = (__bf16)a0.y; f[2]  = (__bf16)a0.z; f[3]  = (__bf16)a0.w;
    f[4]  = (__bf16)a1.x; f[5]  = (__bf16)a1.y; f[6]  = (__bf16)a1.z; f[7]  = (__bf16)a1.w;
    f[8]  = (__bf16)a2.x; f[9]  = (__bf16)a2.y; f[10] = (__bf16)a2.z; f[11] = (__bf16)a2.w;
    f[12] = (__bf16)a3.x; f[13] = (__bf16)a3.y; f[14] = (__bf16)a3.z; f[15] = (__bf16)a3.w;
    Af[kt] = f;
  }

  // ---- Stage B: H = leaky_relu(A @ W1 + b1), kept in LDS as bf16 ----
#pragma unroll 1
  for (int grp = 0; grp < 4; ++grp) {
    __syncthreads();
#if HAS_TDM
    if (w == 0) {  // TDM DMA: 48 contiguous tiles (nt = 8g..8g+7, all kt) = 48 KB
      tdm_stage(wstage_lds_off, (const char*)w1s + grp * 48 * 1024, 48 * 1024);
      __builtin_amdgcn_s_wait_tensorcnt(0);
    }
#else
    {
      const float4* src = (const float4*)((const char*)w1s + grp * 48 * 1024);
      float4* dst = (float4*)Wstage;
#pragma unroll
      for (int i = 0; i < 12; ++i) dst[tid + 256 * i] = src[tid + 256 * i];
    }
#endif
    __syncthreads();
#pragma unroll 1
    for (int j = 0; j < 8; ++j) {
      const int nt = grp * 8 + j;
      const float bv = b1[nt * 16 + lm];
      v8f acc;
#pragma unroll
      for (int r = 0; r < 8; ++r) acc[r] = bv;
#pragma unroll
      for (int kt = 0; kt < 6; ++kt) {
        const __bf16* bp = (const __bf16*)(Wstage + (j * 6 + kt) * 1024) + lane * 16;
        v16bf Bf = concat8(*(const v8bf*)bp, *(const v8bf*)(bp + 8));
        acc = __builtin_amdgcn_wmma_f32_16x16x32_bf16(
            false, Af[kt], false, Bf, (short)0, acc, false, false);
      }
#pragma unroll
      for (int r = 0; r < 8; ++r) {
        const float f = fmaxf(acc[r], 0.01f * acc[r]);  // leaky_relu
        const int m = m_base + r + half * 8;
        Hs[m * HS_STRIDE + nt * 16 + lm] = (__bf16)f;
      }
    }
  }

  // ---- Stage C: proj = H @ W2 + b2 (16 resident f32 accumulators) ----
  v8f acc2[16];
#pragma unroll
  for (int nt = 0; nt < 16; ++nt) {
    const float bv = b2[nt * 16 + lm];
#pragma unroll
    for (int r = 0; r < 8; ++r) acc2[nt][r] = bv;
  }

#pragma unroll 1
  for (int kt = 0; kt < 16; ++kt) {
    __syncthreads();
#if HAS_TDM
    if (w == 0) {  // one K-slab of W2: tiles (kt, nt=0..15) contiguous = 16 KB
      tdm_stage(wstage_lds_off, (const char*)w2s + kt * 16 * 1024, 16 * 1024);
      __builtin_amdgcn_s_wait_tensorcnt(0);
    }
#else
    {
      const float4* src = (const float4*)((const char*)w2s + kt * 16 * 1024);
      float4* dst = (float4*)Wstage;
#pragma unroll
      for (int i = 0; i < 4; ++i) dst[tid + 256 * i] = src[tid + 256 * i];
    }
#endif
    __syncthreads();
    // H A-fragment for this kt (own slab; two ds_load_b128)
    const __bf16* hp = Hs + (m_base + lm) * HS_STRIDE + kt * 32 + klo;
    v16bf Hf = concat8(*(const v8bf*)hp, *(const v8bf*)(hp + 8));
#pragma unroll
    for (int nt = 0; nt < 16; ++nt) {
      const __bf16* bp = (const __bf16*)(Wstage + nt * 1024) + lane * 16;
      v16bf Bf = concat8(*(const v8bf*)bp, *(const v8bf*)(bp + 8));
      acc2[nt] = __builtin_amdgcn_wmma_f32_16x16x32_bf16(
          false, Hf, false, Bf, (short)0, acc2[nt], false, false);
    }
  }

  // ---- Write proj with branch split: out[(1+br)*NE*NF + e*NF + f] ----
#pragma unroll
  for (int nt = 0; nt < 16; ++nt) {
    const int n2 = nt * 16 + lm;
    const int br = n2 >> 7;
    const int f  = n2 & 127;
    const size_t obase = (size_t)(1 + br) * NE * NF + (size_t)f;
#pragma unroll
    for (int r = 0; r < 8; ++r) {
      const int e = blockIdx.x * 128 + m_base + r + half * 8;
      out[obase + (size_t)e * NF] = acc2[nt][r];
    }
  }
}

// ---------------------------------------------------------------------------
extern "C" void kernel_launch(void* const* d_in, const int* in_sizes, int n_in,
                              void* d_out, int out_size, void* d_ws, size_t ws_size,
                              hipStream_t stream) {
  const float* x  = (const float*)d_in[0];
  const float* g  = (const float*)d_in[1];
  const float* W1 = (const float*)d_in[2];
  const float* b1 = (const float*)d_in[3];
  const float* W2 = (const float*)d_in[4];
  const float* b2 = (const float*)d_in[5];
  const int* pidx = (const int*)d_in[6];
  float* out = (float*)d_out;

  v2bf* w1s = (v2bf*)d_ws;
  v2bf* w2s = (v2bf*)((char*)d_ws + W1S_BYTES);

  (void)hipFuncSetAttribute((const void*)fused_branching,
                            hipFuncAttributeMaxDynamicSharedMemorySize, SMEM_BYTES);

  // 1) weights fp32 -> swizzled bf16
  prep_weights<<<(W1_DW + W2_DW + 255) / 256, 256, 0, stream>>>(W1, W2, w1s, w2s);
  // 2) passthrough copy of x into out[0 : NE*NF]
  copy_x<<<(NE * NF / 4) / 256, 256, 0, stream>>>((const float4*)x, (float4*)out);
  // 3) fused branching MLP
  fused_branching<<<NE / 128, 256, SMEM_BYTES, stream>>>(
      x, g, b1, b2, (const __bf16*)w1s, (const __bf16*)w2s, pidx, out);
}
